// MultiHeadSelfAttentionModel_11433202942282
// MI455X (gfx1250) — matrix operand order repaired and checked by reading, not compiled, not run on hardware
//
#include <hip/hip_runtime.h>
#include <hip/hip_bf16.h>
#include <stdint.h>

// ---------------------------------------------------------------------------
// Problem constants (from reference)
// ---------------------------------------------------------------------------
#define DMODEL 1024
#define SEQ    2048
#define BATCH  4
#define NHEAD  16
#define HEADD  64                 // DMODEL / NHEAD
#define BH     (BATCH * NHEAD)    // 64 flat "heads" after the non-standard reshape
#define MROWS  (BATCH * SEQ)      // 8192 activation rows
#define LNEPS  1e-5f

typedef __attribute__((ext_vector_type(16))) __bf16 v16bf;
typedef __attribute__((ext_vector_type(8)))  __bf16 v8bf;
typedef __attribute__((ext_vector_type(8)))  float  v8f;

union Frag16 { v16bf v; v8bf h[2]; };

static __device__ __forceinline__ __bf16 f2bf(float f) {
    union { float f; uint32_t u; } c; c.f = f;
    uint32_t u = c.u;
    uint32_t r = (u + 0x7FFFu + ((u >> 16) & 1u)) >> 16;
    union { unsigned short s; __bf16 b; } o; o.s = (unsigned short)r;
    return o.b;
}

static __device__ __forceinline__ v8f v8f_zero() {
    v8f z;
#pragma unroll
    for (int i = 0; i < 8; ++i) z[i] = 0.0f;
    return z;
}

static __device__ __forceinline__ v8f wmma_bf16(v16bf a, v16bf b, v8f c) {
    return __builtin_amdgcn_wmma_f32_16x16x32_bf16(false, a, false, b,
                                                   (short)0, c, false, false);
}

// ---------------------------------------------------------------------------
// 1) f32 -> bf16 conversion (weights)
// ---------------------------------------------------------------------------
__global__ void cvt_bf16_kernel(const float* __restrict__ in,
                                __bf16* __restrict__ out, int n) {
    int i = blockIdx.x * blockDim.x + threadIdx.x;
    if (i < n) out[i] = f2bf(in[i]);
}

// ---------------------------------------------------------------------------
// 2) Embedding gather: x f32 + bf16 copy for WMMA
// ---------------------------------------------------------------------------
__global__ void gather_kernel(const int* __restrict__ inp,
                              const float* __restrict__ emb,
                              float* __restrict__ x,
                              __bf16* __restrict__ xh) {
    size_t i = (size_t)blockIdx.x * blockDim.x + threadIdx.x;  // < MROWS*DMODEL
    int row = (int)(i >> 10);
    int col = (int)(i & (DMODEL - 1));
    int tok = inp[row];
    float v = emb[(size_t)tok * DMODEL + col];
    x[i]  = v;
    xh[i] = f2bf(v);
}

// ---------------------------------------------------------------------------
// 3) GEMM  C[M,N] = A[M,K] * W[N,K]^T   (NT, both K-contiguous)
//    mode 0: store bf16 (Q/K/V).  mode 1: f32 = acc + bias[col] + resid (WO).
//    Block = 4 waves; each wave owns a 32x64 output tile: 2 M-subtiles x
//    4 N-subtiles, so every B fragment is reused by two WMMAs.
// ---------------------------------------------------------------------------
__global__ void gemm_nt_kernel(const __bf16* __restrict__ A,
                               const __bf16* __restrict__ W,
                               __bf16* __restrict__ outb,
                               float* __restrict__ outf,
                               const float* __restrict__ bias,
                               const float* __restrict__ resid,
                               int mode) {
    const int wave = threadIdx.x >> 5;
    const int lane = threadIdx.x & 31;
    const int half = lane >> 4;        // A/B: K-half select / C: +8 row select
    const int mr   = lane & 15;
    const int m0   = blockIdx.x * 128 + wave * 32;
    const int n0   = blockIdx.y * 64;

    v8f acc[2][4];
#pragma unroll
    for (int mi = 0; mi < 2; ++mi)
#pragma unroll
        for (int nb = 0; nb < 4; ++nb) acc[mi][nb] = v8f_zero();

    const __bf16* arow0 = A + (size_t)(m0 + mr) * DMODEL;
    const __bf16* arow1 = A + (size_t)(m0 + 16 + mr) * DMODEL;

    for (int k0 = 0; k0 < DMODEL; k0 += 32) {
        __builtin_prefetch(arow0 + k0 + 256, 0, 3);   // global_prefetch_b8 (near)
        __builtin_prefetch(arow1 + k0 + 256, 0, 3);
        Frag16 a0, a1;                                 // two A 16x32 fragments
        a0.h[0] = *(const v8bf*)(arow0 + k0 + (half ? 8  : 0));
        a0.h[1] = *(const v8bf*)(arow0 + k0 + (half ? 24 : 16));
        a1.h[0] = *(const v8bf*)(arow1 + k0 + (half ? 8  : 0));
        a1.h[1] = *(const v8bf*)(arow1 + k0 + (half ? 24 : 16));
#pragma unroll
        for (int nb = 0; nb < 4; ++nb) {               // B 32x16 fragment, reused 2x
            const __bf16* wrow =
                W + (size_t)(n0 + nb * 16 + mr) * DMODEL + k0 + half * 8;
            Frag16 bfr;
            bfr.h[0] = *(const v8bf*)(wrow);
            bfr.h[1] = *(const v8bf*)(wrow + 16);
            acc[0][nb] = wmma_bf16(a0.v, bfr.v, acc[0][nb]);
            acc[1][nb] = wmma_bf16(a1.v, bfr.v, acc[1][nb]);
        }
    }

#pragma unroll
    for (int mi = 0; mi < 2; ++mi) {
#pragma unroll
        for (int nb = 0; nb < 4; ++nb) {
#pragma unroll
            for (int r = 0; r < 8; ++r) {
                int row = m0 + mi * 16 + r + 8 * half;
                int col = n0 + nb * 16 + mr;
                size_t idx = (size_t)row * DMODEL + col;
                float v = acc[mi][nb][r];
                if (mode == 1) outf[idx] = v + bias[col] + resid[idx];
                else           outb[idx] = f2bf(v);
            }
        }
    }
}

// ---------------------------------------------------------------------------
// 4) V transpose per flat head:  Vt[bh][e][n] = V[bh][n][e]
// ---------------------------------------------------------------------------
__global__ void transpose_v_kernel(const __bf16* __restrict__ vh,
                                   __bf16* __restrict__ vt) {
    size_t i = (size_t)blockIdx.x * blockDim.x + threadIdx.x;  // < BH*SEQ*HEADD
    int bh  = (int)(i >> 17);
    int rem = (int)(i & 131071);
    int n   = rem >> 6;
    int e   = rem & 63;
    vt[((size_t)bh * HEADD + e) * SEQ + n] = vh[i];
}

// ---------------------------------------------------------------------------
// 5) Flash attention: 1 wave per 16 Q rows; online softmax; 8 wmma / 32 keys.
//    grid = (SEQ/64, BH), block = 128 (4 waves).
// ---------------------------------------------------------------------------
__global__ void flash_attn_kernel(const __bf16* __restrict__ Q,
                                  const __bf16* __restrict__ Km,
                                  const __bf16* __restrict__ Vt,
                                  __bf16* __restrict__ Outp) {
    __shared__ __align__(16) __bf16 plds[4][16][32];  // per-wave P tile (bf16)

    const int wave = threadIdx.x >> 5;
    const int lane = threadIdx.x & 31;
    const int half = lane >> 4;
    const int mr   = lane & 15;
    const int bh   = blockIdx.y;
    const int q0   = (blockIdx.x * 4 + wave) * 16;

    // Q A-fragments for the two K=32 steps over d = 0..63 (resident all loop).
    const __bf16* qrow = Q + ((size_t)bh * SEQ + q0 + mr) * HEADD;
    Frag16 qa0, qa1;
    qa0.h[0] = *(const v8bf*)(qrow +      (half ? 8  : 0));
    qa0.h[1] = *(const v8bf*)(qrow +      (half ? 24 : 16));
    qa1.h[0] = *(const v8bf*)(qrow + 32 + (half ? 8  : 0));
    qa1.h[1] = *(const v8bf*)(qrow + 32 + (half ? 24 : 16));

    float mst[8], lst[8];
#pragma unroll
    for (int r = 0; r < 8; ++r) { mst[r] = -1e30f; lst[r] = 0.0f; }
    v8f acc[4];
#pragma unroll
    for (int e = 0; e < 4; ++e) acc[e] = v8f_zero();

    const __bf16* kbase = Km + (size_t)bh * SEQ * HEADD;
    const __bf16* vbase = Vt + (size_t)bh * HEADD * SEQ;

    for (int n0 = 0; n0 < SEQ; n0 += 32) {
        // ---- scores: two 16x16 tiles covering 32 keys -----------------
        v8f s0 = v8f_zero(), s1 = v8f_zero();
        {
            const __bf16* kr = kbase + (size_t)(n0 + mr) * HEADD + half * 8;
            Frag16 kb;
            kb.h[0] = *(const v8bf*)(kr);      kb.h[1] = *(const v8bf*)(kr + 16);
            s0 = wmma_bf16(qa0.v, kb.v, s0);
            kb.h[0] = *(const v8bf*)(kr + 32); kb.h[1] = *(const v8bf*)(kr + 48);
            s0 = wmma_bf16(qa1.v, kb.v, s0);
            const __bf16* kr2 = kbase + (size_t)(n0 + 16 + mr) * HEADD + half * 8;
            kb.h[0] = *(const v8bf*)(kr2);      kb.h[1] = *(const v8bf*)(kr2 + 16);
            s1 = wmma_bf16(qa0.v, kb.v, s1);
            kb.h[0] = *(const v8bf*)(kr2 + 32); kb.h[1] = *(const v8bf*)(kr2 + 48);
            s1 = wmma_bf16(qa1.v, kb.v, s1);
        }

        // ---- online softmax (rows live in lane-halves of the C layout) --
        float pr0[8], pr1[8], alpha[8];
#pragma unroll
        for (int r = 0; r < 8; ++r) {
            float x0 = s0[r] * 0.125f;   // 1/sqrt(64)
            float x1 = s1[r] * 0.125f;
            float cm = fmaxf(x0, x1);
#pragma unroll
            for (int off = 1; off < 16; off <<= 1)
                cm = fmaxf(cm, __shfl_xor(cm, off, 32));
            float nm = fmaxf(mst[r], cm);
            float al = __expf(mst[r] - nm);
            float p0 = __expf(x0 - nm);
            float p1 = __expf(x1 - nm);
            float rs = p0 + p1;
#pragma unroll
            for (int off = 1; off < 16; off <<= 1)
                rs += __shfl_xor(rs, off, 32);
            lst[r] = lst[r] * al + rs;
            mst[r] = nm;
            pr0[r] = p0; pr1[r] = p1; alpha[r] = al;
        }
#pragma unroll
        for (int e = 0; e < 4; ++e)
#pragma unroll
            for (int r = 0; r < 8; ++r) acc[e][r] *= alpha[r];

        // ---- C-layout -> A-layout for P via per-wave LDS tile -----------
        __syncthreads();
#pragma unroll
        for (int r = 0; r < 8; ++r) {
            int row = r + 8 * half;
            plds[wave][row][mr]      = f2bf(pr0[r]);
            plds[wave][row][16 + mr] = f2bf(pr1[r]);
        }
        __syncthreads();
        Frag16 pa;
        pa.h[0] = *(const v8bf*)&plds[wave][mr][half ? 8  : 0];
        pa.h[1] = *(const v8bf*)&plds[wave][mr][half ? 24 : 16];

        // ---- acc += P (16x32) * V (32x64), V^T is K-contiguous ----------
#pragma unroll
        for (int e = 0; e < 4; ++e) {
            const __bf16* vr = vbase + (size_t)(e * 16 + mr) * SEQ + n0 + half * 8;
            Frag16 vb;
            vb.h[0] = *(const v8bf*)(vr);
            vb.h[1] = *(const v8bf*)(vr + 16);
            acc[e] = wmma_bf16(pa.v, vb.v, acc[e]);
        }
    }

    // ---- epilogue: normalize by row sums, store attended (bf16) ---------
#pragma unroll
    for (int r = 0; r < 8; ++r) {
        float inv = 1.0f / lst[r];
        int row = q0 + r + 8 * half;
#pragma unroll
        for (int e = 0; e < 4; ++e)
            Outp[((size_t)bh * SEQ + row) * HEADD + e * 16 + mr] =
                f2bf(acc[e][r] * inv);
    }
}

// ---------------------------------------------------------------------------
// 6) LayerNorm fused with the wf dot: rowdot[row] = ln(h_row) . wf
// ---------------------------------------------------------------------------
__global__ void lnorm_dot_kernel(const float* __restrict__ h,
                                 const float* __restrict__ gamma,
                                 const float* __restrict__ beta,
                                 const float* __restrict__ wf,
                                 float* __restrict__ rowdot) {
    const int row = blockIdx.x;
    const int tid = threadIdx.x;
    const float* hr = h + (size_t)row * DMODEL;

    __shared__ float r1[256], r2[256];
    float s = 0.0f, s2 = 0.0f;
    for (int c = tid; c < DMODEL; c += 256) { float v = hr[c]; s += v; s2 += v * v; }
    r1[tid] = s; r2[tid] = s2;
    __syncthreads();
    for (int o = 128; o > 0; o >>= 1) {
        if (tid < o) { r1[tid] += r1[tid + o]; r2[tid] += r2[tid + o]; }
        __syncthreads();
    }
    float mu  = r1[0] * (1.0f / DMODEL);
    float var = r2[0] * (1.0f / DMODEL) - mu * mu;
    float inv = rsqrtf(var + LNEPS);
    __syncthreads();

    float d = 0.0f;
    for (int c = tid; c < DMODEL; c += 256)
        d += ((hr[c] - mu) * inv * gamma[c] + beta[c]) * wf[c];
    r1[tid] = d;
    __syncthreads();
    for (int o = 128; o > 0; o >>= 1) {
        if (tid < o) r1[tid] += r1[tid + o];
        __syncthreads();
    }
    if (tid == 0) rowdot[row] = r1[0];
}

// ---------------------------------------------------------------------------
// 7) Deterministic final reduce: out[b] = mean_s(rowdot) + bf
// ---------------------------------------------------------------------------
__global__ void finalize_kernel(const float* __restrict__ rowdot,
                                const float* __restrict__ bfp,
                                float* __restrict__ out) {
    const int b = blockIdx.x;
    const int tid = threadIdx.x;
    const float* rp = rowdot + (size_t)b * SEQ;
    __shared__ float r[256];
    float s = 0.0f;
    for (int i = tid; i < SEQ; i += 256) s += rp[i];
    r[tid] = s;
    __syncthreads();
    for (int o = 128; o > 0; o >>= 1) {
        if (tid < o) r[tid] += r[tid + o];
        __syncthreads();
    }
    if (tid == 0) out[b] = r[0] * (1.0f / SEQ) + bfp[0];
}

// ---------------------------------------------------------------------------
// Launch
// ---------------------------------------------------------------------------
extern "C" void kernel_launch(void* const* d_in, const int* in_sizes, int n_in,
                              void* d_out, int out_size, void* d_ws, size_t ws_size,
                              hipStream_t stream) {
    (void)in_sizes; (void)n_in; (void)out_size; (void)ws_size;

    const int*   inp   = (const int*)  d_in[0];
    const float* emb   = (const float*)d_in[1];
    const float* wq    = (const float*)d_in[2];
    const float* wk    = (const float*)d_in[3];
    const float* wv    = (const float*)d_in[4];
    const float* wo    = (const float*)d_in[5];
    const float* bo    = (const float*)d_in[6];
    const float* gamma = (const float*)d_in[7];
    const float* beta  = (const float*)d_in[8];
    const float* wf    = (const float*)d_in[9];
    const float* bfp   = (const float*)d_in[10];
    float* out = (float*)d_out;

    const size_t nAct = (size_t)MROWS * DMODEL;          // 8M elems
    const size_t nW   = (size_t)DMODEL * DMODEL;         // 1M elems

    // workspace carve-up (bytes)
    char* ws = (char*)d_ws;
    float*  x    = (float*) (ws);                                 // 32 MB f32
    __bf16* xh   = (__bf16*)(ws + nAct * 4);                      // 16 MB bf16 (reused as attended)
    __bf16* wqh  = (__bf16*)(ws + nAct * 6);                      // 4 x 2 MB bf16 weights
    __bf16* wkh  = wqh + nW;
    __bf16* wvh  = wkh + nW;
    __bf16* woh  = wvh + nW;
    __bf16* qh   = woh + nW;                                      // 16 MB
    __bf16* kh   = qh  + nAct;                                    // 16 MB
    __bf16* vh   = kh  + nAct;                                    // 16 MB
    __bf16* vt   = vh  + nAct;                                    // 16 MB
    float*  hbuf = (float*)(vt + nAct);                           // 32 MB
    float*  rowdot = hbuf + nAct;                                 // 32 KB

    // 1) weights -> bf16
    cvt_bf16_kernel<<<(int)(nW / 256), 256, 0, stream>>>(wq, wqh, (int)nW);
    cvt_bf16_kernel<<<(int)(nW / 256), 256, 0, stream>>>(wk, wkh, (int)nW);
    cvt_bf16_kernel<<<(int)(nW / 256), 256, 0, stream>>>(wv, wvh, (int)nW);
    cvt_bf16_kernel<<<(int)(nW / 256), 256, 0, stream>>>(wo, woh, (int)nW);

    // 2) embedding gather
    gather_kernel<<<(int)(nAct / 256), 256, 0, stream>>>(inp, emb, x, xh);

    // 3) Q/K/V projections (WMMA)
    dim3 ggrid(MROWS / 128, DMODEL / 64);
    gemm_nt_kernel<<<ggrid, 128, 0, stream>>>(xh, wqh, qh, nullptr, nullptr, nullptr, 0);
    gemm_nt_kernel<<<ggrid, 128, 0, stream>>>(xh, wkh, kh, nullptr, nullptr, nullptr, 0);
    gemm_nt_kernel<<<ggrid, 128, 0, stream>>>(xh, wvh, vh, nullptr, nullptr, nullptr, 0);

    // 4) V transpose per head
    transpose_v_kernel<<<(int)(nAct / 256), 256, 0, stream>>>(vh, vt);

    // 5) flash attention -> attended (reuse xh buffer; QKV GEMMs done)
    __bf16* attended = xh;
    flash_attn_kernel<<<dim3(SEQ / 64, BH), 128, 0, stream>>>(qh, kh, vt, attended);

    // 6) output projection + bias + residual (WMMA, f32 out)
    gemm_nt_kernel<<<ggrid, 128, 0, stream>>>(attended, woh, nullptr, hbuf, bo, x, 1);

    // 7) LayerNorm fused with wf dot
    lnorm_dot_kernel<<<MROWS, 256, 0, stream>>>(hbuf, gamma, beta, wf, rowdot);

    // 8) deterministic pooled logits
    finalize_kernel<<<BATCH, 256, 0, stream>>>(rowdot, bfp, out);
}
